// ViTSeg_42202348650874
// MI455X (gfx1250) — matrix operand order repaired and checked by reading, not compiled, not run on hardware
//
#include <hip/hip_runtime.h>
#include <cstdint>
#include <cstddef>

typedef __bf16 bf16_t;
typedef __attribute__((ext_vector_type(16))) __bf16 v16bf;
typedef __attribute__((ext_vector_type(8)))  __bf16 v8bf;
typedef __attribute__((ext_vector_type(8)))  float  v8f;
typedef __attribute__((ext_vector_type(4)))  int    v4i_t;

#define BM 64
#define BN 128
#define BK 32

// ---- optional CDNA5 async global->LDS copy (ASYNCcnt-tracked) --------------
#if defined(__has_builtin)
#  if __has_builtin(__builtin_amdgcn_global_load_async_to_lds_b128)
#    define USE_ASYNC_CP 1
#  endif
#endif
#ifndef USE_ASYNC_CP
#  define USE_ASYNC_CP 0
#endif

#if USE_ASYNC_CP
typedef __attribute__((address_space(1))) v4i_t* g_v4i_p;
typedef __attribute__((address_space(3))) void*  l_void_p;
typedef __attribute__((address_space(3))) v4i_t* l_v4i_p;
__device__ __forceinline__ void cp_async16(const void* g, void* l) {
  // global addresses are identity-mapped in the flat aperture -> int round-trip
  // is exact for AS(1); LDS needs a real addrspacecast (generic -> AS(3)).
  l_void_p l3 = (l_void_p)l;            // addrspacecast generic -> local
  __builtin_amdgcn_global_load_async_to_lds_b128(
      (g_v4i_p)(unsigned long long)g,   // inttoptr into AS(1), exact for global
      (l_v4i_p)l3,                      // pointee reinterpret within AS(3)
      0, 0);
}
__device__ __forceinline__ void cp_async_wait() {
  asm volatile("s_wait_asynccnt 0" ::: "memory");
}
#endif

// ---------------------------------------------------------------------------
// WMMA bf16 GEMM: C[M,N] = A[M,K] * B^T  where B is stored [N,K] row-major.
//   Batched via blockIdx.z with element strides sA/sB/sC.
//   Requires: M % 64 == 0, K % 32 == 0, lda/ldb multiples of 8, 16B-aligned
//   bases.  B rows may be over-read up to n0+127 (caller guarantees the
//   memory behind B has >= (ceil(N/128)*128)*ldb elements reachable);
//   out-of-range columns are never stored.
// EPI: 0=none 1=+bias 2=+bias,gelu 3=+bias,+residual
// ---------------------------------------------------------------------------
template<int EPI>
__global__ __launch_bounds__(256)
void k_gemm(const bf16_t* __restrict__ A, const bf16_t* __restrict__ Bm,
            float* __restrict__ C, bf16_t* __restrict__ Cbf,
            const float* __restrict__ bias, const float* __restrict__ resid,
            int M, int N, int K, int lda, int ldb, int ldc,
            long long sA, long long sB, long long sC)
{
  __shared__ __align__(16) bf16_t As[BM][BK + 8];   // [m][k]
  __shared__ __align__(16) bf16_t Bs[BN][BK + 8];   // [n][k]

  const int z = blockIdx.z;
  const bf16_t* Ab = A  + (size_t)z * (size_t)sA;
  const bf16_t* Bb = Bm + (size_t)z * (size_t)sB;
  const int m0 = blockIdx.y * BM;
  const int n0 = blockIdx.x * BN;
  const int t    = threadIdx.x;
  const int lane = t & 31;
  const int wid  = t >> 5;
  const int wm   = (wid >> 2) * 32;   // 0 or 32
  const int wn   = (wid & 3) * 32;    // 0,32,64,96

  const v8f vzero = {0.f,0.f,0.f,0.f,0.f,0.f,0.f,0.f};
  v8f acc[2][2];
  acc[0][0] = vzero; acc[0][1] = vzero; acc[1][0] = vzero; acc[1][1] = vzero;

  const int mrow = lane & 15;       // row within 16x16 tile (A) / col (B)
  const int kh   = lane >> 4;       // K-half selector

  // per-thread staging coords (compile-time shapes, no guards)
  const int ar = t >> 2, aseg = t & 3;            // A: 64 rows x 4 segs
  const bf16_t* asrc0 = Ab + (size_t)(m0 + ar) * lda + aseg * 8;

  for (int k0 = 0; k0 < K; k0 += BK) {
    // ---- stage A tile: 64x32, 8 bf16 (16B) per thread, contiguous ----
    {
      const bf16_t* src = asrc0 + k0;
#if USE_ASYNC_CP
      cp_async16(src, &As[ar][aseg * 8]);
#else
      *(uint4*)&As[ar][aseg * 8] = *(const uint4*)src;
#endif
      if (k0 + BK < K)
        __builtin_prefetch(src + BK, 0, 0);
    }
    // ---- stage B tile: 128x32, two 16B segments per thread ----
    for (int it = 0; it < 2; ++it) {
      const int s = t + it * 256;
      const int n = s >> 2, seg = s & 3;
      const bf16_t* src = Bb + (size_t)(n0 + n) * ldb + (k0 + seg * 8);
#if USE_ASYNC_CP
      cp_async16(src, &Bs[n][seg * 8]);
#else
      *(uint4*)&Bs[n][seg * 8] = *(const uint4*)src;
#endif
    }
#if USE_ASYNC_CP
    cp_async_wait();
#endif
    __syncthreads();

    // ---- fragments (ISA 16-bit A 16x32 / B 32x16 striping) ----
    union Frag { v16bf v; v8bf h[2]; };
    Frag a[2], b[2];
    for (int i = 0; i < 2; ++i) {
      const int mr = wm + i * 16 + mrow;
      a[i].h[0] = *(const v8bf*)&As[mr][kh * 8];        // K = 8*kh + 0..7
      a[i].h[1] = *(const v8bf*)&As[mr][kh * 8 + 16];   // K = 8*kh + 16..23
      const int nr = wn + i * 16 + mrow;
      b[i].h[0] = *(const v8bf*)&Bs[nr][kh * 8];        // K = 8*kh + 0..7
      b[i].h[1] = *(const v8bf*)&Bs[nr][kh * 8 + 16];   // K = 8*kh + 16..23
    }
    for (int i = 0; i < 2; ++i)
      for (int j = 0; j < 2; ++j)
        acc[i][j] = __builtin_amdgcn_wmma_f32_16x16x32_bf16(
            false, a[i].v, false, b[j].v, (short)0, acc[i][j], false, false);
    __syncthreads();
  }

  // ---- epilogue ----
  for (int i = 0; i < 2; ++i) {
    for (int j = 0; j < 2; ++j) {
      const int col = n0 + wn + j * 16 + (lane & 15);
      if (col >= N) continue;
      const int rbase = m0 + wm + i * 16 + ((lane >> 4) * 8);
      const float bb = (EPI >= 1) ? bias[col] : 0.f;
      for (int r = 0; r < 8; ++r) {
        const int row = rbase + r;
        float v = acc[i][j][r] + bb;
        if (EPI == 2) v = 0.5f * v * (1.f + erff(v * 0.70710678118f));
        if (EPI == 3) v += resid[(size_t)row * ldc + col];
        const size_t idx = (size_t)z * (size_t)sC + (size_t)row * ldc + col;
        if (C)   C[idx]   = v;
        if (Cbf) Cbf[idx] = (bf16_t)v;
      }
    }
  }
}

// ---------------------------------------------------------------------------
// Elementwise / helper kernels
// ---------------------------------------------------------------------------

// fp32 [L,K,N] -> bf16 transposed per slice: out[l][n][k]
__global__ void k_cvt_t(const float* __restrict__ in, bf16_t* __restrict__ out,
                        int Kd, int Nd, int L) {
  size_t i = (size_t)blockIdx.x * blockDim.x + threadIdx.x;
  const size_t per = (size_t)Kd * Nd;
  const size_t total = per * L;
  if (i >= total) return;
  int l = (int)(i / per);
  size_t r = i % per;
  int k = (int)(r / Nd), n = (int)(r % Nd);
  out[(size_t)l * per + (size_t)n * Kd + k] = (bf16_t)in[i];
}

// img [16,3,384,384] -> patches bf16 [9216,768], pd = (p1*16+p2)*3 + c
__global__ void k_patchify(const float* __restrict__ img, bf16_t* __restrict__ patches) {
  size_t i = (size_t)blockIdx.x * blockDim.x + threadIdx.x;
  const size_t total = (size_t)9216 * 768;
  if (i >= total) return;
  int pd = (int)(i % 768);
  size_t row = i / 768;
  int b = (int)(row / 576), n = (int)(row % 576);
  int hy = n / 24, wx = n % 24;
  int p1 = pd / 48; int rem = pd % 48; int p2 = rem / 3; int c = rem % 3;
  int yy = hy * 16 + p1, xx = wx * 16 + p2;
  patches[i] = (bf16_t)img[(((size_t)b * 3 + c) * 384 + yy) * 384 + xx];
}

__global__ void k_add_pos(float* __restrict__ x, const float* __restrict__ pos, size_t total) {
  size_t i = (size_t)blockIdx.x * blockDim.x + threadIdx.x;
  if (i < total) x[i] += pos[i % (size_t)(576 * 256)];
}

// LayerNorm over D=256, one block (256 thr) per row; bf16 and/or f32 outputs
__global__ __launch_bounds__(256)
void k_layernorm(const float* __restrict__ x, const float* __restrict__ g,
                 const float* __restrict__ b, bf16_t* __restrict__ obf,
                 float* __restrict__ of32) {
  const int row = blockIdx.x, t = threadIdx.x;
  const float v = x[(size_t)row * 256 + t];
  __shared__ float s1[256], s2[256];
  s1[t] = v; s2[t] = v * v;
  __syncthreads();
  for (int s = 128; s > 0; s >>= 1) {
    if (t < s) { s1[t] += s1[t + s]; s2[t] += s2[t + s]; }
    __syncthreads();
  }
  const float mean = s1[0] * (1.f / 256.f);
  const float var  = s2[0] * (1.f / 256.f) - mean * mean;
  const float y = (v - mean) * rsqrtf(var + 1e-5f) * g[t] + b[t];
  if (obf)  obf[(size_t)row * 256 + t]  = (bf16_t)y;
  if (of32) of32[(size_t)row * 256 + t] = y;
}

// qkv fp32 [B,N,768] -> q(*scale) [bh,N,64], k [bh,N,64], v^T [bh,64,N] bf16
__global__ void k_split_qkv(const float* __restrict__ qkv, bf16_t* __restrict__ qs,
                            bf16_t* __restrict__ ks, bf16_t* __restrict__ vst) {
  size_t i = (size_t)blockIdx.x * blockDim.x + threadIdx.x;
  const size_t total = (size_t)9216 * 768;
  if (i >= total) return;
  int c = (int)(i % 768);
  size_t bn = i / 768;
  int which = c >> 8; int within = c & 255;
  int h = within >> 6, d = within & 63;
  int b = (int)(bn / 576), n = (int)(bn % 576);
  float v = qkv[i];
  if (which == 0) {
    qs[((size_t)(b * 4 + h) * 576 + n) * 64 + d] = (bf16_t)(v * 0.125f); // DH^-0.5
  } else if (which == 1) {
    ks[((size_t)(b * 4 + h) * 576 + n) * 64 + d] = (bf16_t)v;
  } else {
    vst[((size_t)(b * 4 + h) * 64 + d) * 576 + n] = (bf16_t)v;           // transposed
  }
}

// Per-row top-16 + dual softmax mix.  One wave (32 lanes) per row of 576.
__global__ __launch_bounds__(32)
void k_topk_mix(const float* __restrict__ dots, bf16_t* __restrict__ attnp,
                const float* __restrict__ alpha_p, int layer) {
  const int row  = blockIdx.x;                 // 0 .. 64*576-1
  const int lane = threadIdx.x;
  const float* d = dots + (size_t)row * 576;
  bf16_t* o = attnp + (size_t)row * 576;
  __shared__ float orig[576];
  __shared__ float wv[576];
  __shared__ unsigned char sel[576];
  for (int i = lane; i < 576; i += 32) {
    float v = d[i]; orig[i] = v; wv[i] = v; sel[i] = 0;
  }
  __syncthreads();
  float mx = -3.4e38f;
  for (int i = lane; i < 576; i += 32) mx = fmaxf(mx, orig[i]);
  for (int m = 16; m > 0; m >>= 1) mx = fmaxf(mx, __shfl_xor(mx, m, 32));
  float gsum = 0.f;
  for (int i = lane; i < 576; i += 32) gsum += expf(orig[i] - mx);
  for (int m = 16; m > 0; m >>= 1) gsum += __shfl_xor(gsum, m, 32);
  float ssum = 0.f;
  for (int it = 0; it < 16; ++it) {
    float bv = -3.4e38f; int bi = 0;
    for (int i = lane; i < 576; i += 32) {
      float v = wv[i];
      if (v > bv) { bv = v; bi = i; }
    }
    for (int m = 16; m > 0; m >>= 1) {
      float ov = __shfl_xor(bv, m, 32);
      int   oi = __shfl_xor(bi, m, 32);
      if (ov > bv || (ov == bv && oi < bi)) { bv = ov; bi = oi; }
    }
    ssum += expf(bv - mx);
    if (lane == 0) { sel[bi] = 1; wv[bi] = -3.4e38f; }
    __syncthreads();
  }
  float a = alpha_p[layer];
  a = fminf(fmaxf(a, 0.f), 1.f);
  const float ig = a / gsum, is = (1.f - a) / ssum;
  for (int i = lane; i < 576; i += 32) {
    float e = expf(orig[i] - mx);
    o[i] = (bf16_t)(e * ig + (sel[i] ? e * is : 0.f));
  }
}

// per-head out fp32 [B*H,576,64] -> merged bf16 [9216,256]
__global__ void k_merge_heads(const float* __restrict__ oh, bf16_t* __restrict__ merged) {
  size_t i = (size_t)blockIdx.x * blockDim.x + threadIdx.x;
  const size_t total = (size_t)9216 * 256;
  if (i >= total) return;
  int c = (int)(i & 255);
  size_t bn = i >> 8;
  int h = c >> 6, d = c & 63;
  int b = (int)(bn / 576), n = (int)(bn % 576);
  merged[i] = (bf16_t)oh[((size_t)(b * 4 + h) * 576 + n) * 64 + d];
}

// [B,N,D] fp32 -> [B,D,24,24]
__global__ void k_to_nchw(const float* __restrict__ xf, float* __restrict__ t0) {
  size_t i = (size_t)blockIdx.x * blockDim.x + threadIdx.x;
  const size_t total = (size_t)16 * 256 * 24 * 24;
  if (i >= total) return;
  int xx = (int)(i % 24);
  int yy = (int)((i / 24) % 24);
  int dd = (int)((i / 576) % 256);
  int b  = (int)(i / ((size_t)576 * 256));
  t0[i] = xf[((size_t)b * 576 + yy * 24 + xx) * 256 + dd];
}

// naive NCHW conv (ks=1 or 3, SAME), optional relu
__global__ void k_conv(const float* __restrict__ in, const float* __restrict__ w,
                       const float* __restrict__ bias, float* __restrict__ out,
                       int Bn, int Ci, int Co, int H, int W, int ks, int relu) {
  size_t i = (size_t)blockIdx.x * blockDim.x + threadIdx.x;
  const size_t total = (size_t)Bn * Co * H * W;
  if (i >= total) return;
  int x = (int)(i % W);
  int y = (int)((i / W) % H);
  int co = (int)((i / ((size_t)W * H)) % Co);
  int b  = (int)(i / ((size_t)W * H * Co));
  const int pad = ks / 2;
  float acc = bias[co];
  for (int ci = 0; ci < Ci; ++ci) {
    for (int ky = 0; ky < ks; ++ky) {
      int iy = y + ky - pad; if (iy < 0 || iy >= H) continue;
      for (int kx = 0; kx < ks; ++kx) {
        int ix = x + kx - pad; if (ix < 0 || ix >= W) continue;
        acc += in[((size_t)(b * Ci + ci) * H + iy) * W + ix] *
               w[((size_t)(co * Ci + ci) * ks + ky) * ks + kx];
      }
    }
  }
  if (relu) acc = fmaxf(acc, 0.f);
  out[i] = acc;
}

// half-pixel bilinear resize NCHW
__global__ void k_bilinear(const float* __restrict__ in, float* __restrict__ out,
                           int Bn, int C, int Hi, int Wi, int Ho, int Wo) {
  size_t i = (size_t)blockIdx.x * blockDim.x + threadIdx.x;
  const size_t total = (size_t)Bn * C * Ho * Wo;
  if (i >= total) return;
  int x = (int)(i % Wo);
  int y = (int)((i / Wo) % Ho);
  size_t bc = i / ((size_t)Wo * Ho);
  float sy = (y + 0.5f) * (float)Hi / (float)Ho - 0.5f;
  float sx = (x + 0.5f) * (float)Wi / (float)Wo - 0.5f;
  sy = fminf(fmaxf(sy, 0.f), (float)(Hi - 1));
  sx = fminf(fmaxf(sx, 0.f), (float)(Wi - 1));
  int y0 = (int)sy; float fy = sy - y0; int y1 = min(y0 + 1, Hi - 1);
  int x0 = (int)sx; float fx = sx - x0; int x1 = min(x0 + 1, Wi - 1);
  const float* p = in + bc * (size_t)Hi * Wi;
  float v00 = p[(size_t)y0 * Wi + x0], v01 = p[(size_t)y0 * Wi + x1];
  float v10 = p[(size_t)y1 * Wi + x0], v11 = p[(size_t)y1 * Wi + x1];
  float top = v00 + (v01 - v00) * fx;
  float bot = v10 + (v11 - v10) * fx;
  out[i] = top + (bot - top) * fy;
}

// ---------------------------------------------------------------------------
// Host driver
// ---------------------------------------------------------------------------
static inline dim3 ew_grid(size_t n) { return dim3((unsigned)((n + 255) / 256)); }

extern "C" void kernel_launch(void* const* d_in, const int* in_sizes, int n_in,
                              void* d_out, int out_size, void* d_ws, size_t ws_size,
                              hipStream_t stream) {
  (void)in_sizes; (void)n_in; (void)out_size; (void)ws_size;
  const float* img     = (const float*)d_in[0];
  const float* patch_w = (const float*)d_in[1];
  const float* patch_b = (const float*)d_in[2];
  const float* pos_emb = (const float*)d_in[3];
  const float* ln1_g   = (const float*)d_in[4];
  const float* ln1_b   = (const float*)d_in[5];
  const float* qkv_w   = (const float*)d_in[6];
  const float* out_w   = (const float*)d_in[7];
  const float* out_b   = (const float*)d_in[8];
  const float* alpha   = (const float*)d_in[9];
  const float* ln2_g   = (const float*)d_in[10];
  const float* ln2_b   = (const float*)d_in[11];
  const float* ff_w1   = (const float*)d_in[12];
  const float* ff_b1   = (const float*)d_in[13];
  const float* ff_w2   = (const float*)d_in[14];
  const float* ff_b2   = (const float*)d_in[15];
  const float* lnf_g   = (const float*)d_in[16];
  const float* lnf_b   = (const float*)d_in[17];
  const float* conv1_w = (const float*)d_in[18];
  const float* conv1_b = (const float*)d_in[19];
  const float* conv2_w = (const float*)d_in[20];
  const float* conv2_b = (const float*)d_in[21];
  const float* conv3_w = (const float*)d_in[22];
  const float* conv3_b = (const float*)d_in[23];
  float* outp = (float*)d_out;
  char* ws = (char*)d_ws;

  const int ROWS = 9216;   // B*N
  size_t off = 0;
  auto alloc = [&](size_t bytes) { size_t o = off; off = (off + bytes + 255) & ~(size_t)255; return o; };
  // weights stored transposed [N,K] bf16
  const size_t o_wpatch = alloc((size_t)256 * 768 * 2);
  const size_t o_wqkv   = alloc((size_t)6 * 768 * 256 * 2);
  const size_t o_wout   = alloc((size_t)6 * 256 * 256 * 2);
  const size_t o_wff1   = alloc((size_t)6 * 512 * 256 * 2);
  const size_t o_wff2   = alloc((size_t)6 * 256 * 512 * 2);
  const size_t o_x      = alloc((size_t)ROWS * 256 * 4);
  const size_t o_xn     = alloc((size_t)ROWS * 256 * 2);
  const size_t o_qkv    = alloc((size_t)ROWS * 768 * 4);   // also reused as xf fp32
  const size_t o_qs     = alloc((size_t)64 * 576 * 64 * 2);
  const size_t o_ks     = alloc((size_t)64 * 576 * 64 * 2);   // over-read spills into vst (safe)
  const size_t o_vst    = alloc((size_t)64 * 64 * 576 * 2);   // over-read spills into oh (safe)
  const size_t o_oh     = alloc((size_t)64 * 576 * 64 * 4);
  const size_t o_merged = alloc((size_t)ROWS * 256 * 2);
  const size_t o_mlph   = alloc((size_t)ROWS * 512 * 2);
  const size_t o_big    = alloc((size_t)64 * 576 * 576 * 4 + (size_t)64 * 576 * 576 * 2);
  // aliases inside the big region (non-overlapping lifetimes):
  const size_t o_dots    = o_big;                                   // fp32 64x576x576
  const size_t o_attnp   = o_big + (size_t)64 * 576 * 576 * 4;      // bf16 64x576x576
  const size_t o_patches = o_big;                                   // bf16 9216x768 (pre-attention)
  const size_t o_t0 = o_big;                                        // decoder (post-transformer)
  const size_t o_c1 = o_t0 + (size_t)16 * 256 * 24 * 24 * 4;
  const size_t o_u1 = o_c1 + (size_t)16 * 128 * 24 * 24 * 4;
  const size_t o_c2 = o_u1 + (size_t)16 * 128 * 48 * 48 * 4;
  const size_t o_u2 = o_c2 + (size_t)16 * 64 * 48 * 48 * 4;
  const size_t o_c3 = o_u2 + (size_t)16 * 64 * 96 * 96 * 4;

  auto BF = [&](size_t o) { return (bf16_t*)(ws + o); };
  auto FP = [&](size_t o) { return (float*)(ws + o); };

  // --- weight conversion to transposed bf16 [N,K] ---
  k_cvt_t<<<ew_grid((size_t)768 * 256), 256, 0, stream>>>(patch_w, BF(o_wpatch), 768, 256, 1);
  k_cvt_t<<<ew_grid((size_t)6 * 256 * 768), 256, 0, stream>>>(qkv_w, BF(o_wqkv), 256, 768, 6);
  k_cvt_t<<<ew_grid((size_t)6 * 256 * 256), 256, 0, stream>>>(out_w, BF(o_wout), 256, 256, 6);
  k_cvt_t<<<ew_grid((size_t)6 * 256 * 512), 256, 0, stream>>>(ff_w1, BF(o_wff1), 256, 512, 6);
  k_cvt_t<<<ew_grid((size_t)6 * 512 * 256), 256, 0, stream>>>(ff_w2, BF(o_wff2), 512, 256, 6);

  // --- patch embed ---
  k_patchify<<<ew_grid((size_t)ROWS * 768), 256, 0, stream>>>(img, BF(o_patches));
  k_gemm<1><<<dim3(2, ROWS / 64, 1), 256, 0, stream>>>(
      BF(o_patches), BF(o_wpatch), FP(o_x), nullptr, patch_b, nullptr,
      ROWS, 256, 768, 768, 768, 256, 0, 0, 0);
  k_add_pos<<<ew_grid((size_t)ROWS * 256), 256, 0, stream>>>(FP(o_x), pos_emb, (size_t)ROWS * 256);

  // --- transformer layers ---
  for (int L = 0; L < 6; ++L) {
    const bf16_t* wq = BF(o_wqkv) + (size_t)L * 768 * 256;
    const bf16_t* wo = BF(o_wout) + (size_t)L * 256 * 256;
    const bf16_t* w1 = BF(o_wff1) + (size_t)L * 512 * 256;
    const bf16_t* w2 = BF(o_wff2) + (size_t)L * 256 * 512;

    // LN1 -> xn (bf16)
    k_layernorm<<<dim3(ROWS), 256, 0, stream>>>(FP(o_x), ln1_g + L * 256, ln1_b + L * 256,
                                                BF(o_xn), nullptr);
    // QKV GEMM (no bias), fp32 out
    k_gemm<0><<<dim3(6, ROWS / 64, 1), 256, 0, stream>>>(
        BF(o_xn), wq, FP(o_qkv), nullptr, nullptr, nullptr,
        ROWS, 768, 256, 256, 256, 768, 0, 0, 0);
    // split heads (q scaled, v transposed)
    k_split_qkv<<<ew_grid((size_t)ROWS * 768), 256, 0, stream>>>(
        FP(o_qkv), BF(o_qs), BF(o_ks), BF(o_vst));
    // dots = q @ k^T  (batched over 64 (b,h); B stored [N,K] = ks)
    k_gemm<0><<<dim3(5, 576 / 64, 64), 256, 0, stream>>>(
        BF(o_qs), BF(o_ks), FP(o_dots), nullptr, nullptr, nullptr,
        576, 576, 64, 64, 64, 576,
        (long long)576 * 64, (long long)576 * 64, (long long)576 * 576);
    // top-16 + dual softmax mix -> attnp (bf16)
    k_topk_mix<<<dim3(64 * 576), 32, 0, stream>>>(FP(o_dots), BF(o_attnp), alpha, L);
    // out_h = attnp @ v   (B stored [N,K] = v^T [64,576])
    k_gemm<0><<<dim3(1, 576 / 64, 64), 256, 0, stream>>>(
        BF(o_attnp), BF(o_vst), FP(o_oh), nullptr, nullptr, nullptr,
        576, 64, 576, 576, 576, 64,
        (long long)576 * 576, (long long)64 * 576, (long long)576 * 64);
    // merge heads -> bf16
    k_merge_heads<<<ew_grid((size_t)ROWS * 256), 256, 0, stream>>>(FP(o_oh), BF(o_merged));
    // out proj + bias + residual -> x
    k_gemm<3><<<dim3(2, ROWS / 64, 1), 256, 0, stream>>>(
        BF(o_merged), wo, FP(o_x), nullptr, out_b + L * 256, FP(o_x),
        ROWS, 256, 256, 256, 256, 256, 0, 0, 0);
    // LN2 -> xn
    k_layernorm<<<dim3(ROWS), 256, 0, stream>>>(FP(o_x), ln2_g + L * 256, ln2_b + L * 256,
                                                BF(o_xn), nullptr);
    // FF1 + bias + gelu -> mlph (bf16)
    k_gemm<2><<<dim3(4, ROWS / 64, 1), 256, 0, stream>>>(
        BF(o_xn), w1, nullptr, BF(o_mlph), ff_b1 + L * 512, nullptr,
        ROWS, 512, 256, 256, 256, 512, 0, 0, 0);
    // FF2 + bias + residual -> x
    k_gemm<3><<<dim3(2, ROWS / 64, 1), 256, 0, stream>>>(
        BF(o_mlph), w2, FP(o_x), nullptr, ff_b2 + L * 256, FP(o_x),
        ROWS, 256, 512, 512, 512, 256, 0, 0, 0);
  }

  // --- final LN (fp32 out, reuse qkv buffer) + decoder ---
  float* xf = FP(o_qkv);
  k_layernorm<<<dim3(ROWS), 256, 0, stream>>>(FP(o_x), lnf_g, lnf_b, nullptr, xf);
  k_to_nchw<<<ew_grid((size_t)16 * 256 * 576), 256, 0, stream>>>(xf, FP(o_t0));
  k_conv<<<ew_grid((size_t)16 * 128 * 24 * 24), 256, 0, stream>>>(
      FP(o_t0), conv1_w, conv1_b, FP(o_c1), 16, 256, 128, 24, 24, 3, 1);
  k_bilinear<<<ew_grid((size_t)16 * 128 * 48 * 48), 256, 0, stream>>>(
      FP(o_c1), FP(o_u1), 16, 128, 24, 24, 48, 48);
  k_conv<<<ew_grid((size_t)16 * 64 * 48 * 48), 256, 0, stream>>>(
      FP(o_u1), conv2_w, conv2_b, FP(o_c2), 16, 128, 64, 48, 48, 3, 1);
  k_bilinear<<<ew_grid((size_t)16 * 64 * 96 * 96), 256, 0, stream>>>(
      FP(o_c2), FP(o_u2), 16, 64, 48, 48, 96, 96);
  k_conv<<<ew_grid((size_t)16 * 21 * 96 * 96), 256, 0, stream>>>(
      FP(o_u2), conv3_w, conv3_b, FP(o_c3), 16, 64, 21, 96, 96, 1, 0);
  k_bilinear<<<ew_grid((size_t)16 * 21 * 384 * 384), 256, 0, stream>>>(
      FP(o_c3), outp, 16, 21, 96, 96, 384, 384);
}
